// MatchingModel_GATv2Sinkhorn_88098369175734
// MI455X (gfx1250) — matrix operand order repaired and checked by reading, not compiled, not run on hardware
//
#include <hip/hip_runtime.h>
#include <hip/hip_bf16.h>

// ---------------------------------------------------------------------------
// Problem constants (match reference)
// ---------------------------------------------------------------------------
#define BGR   256                 // graphs per batch
#define NPG   256                 // nodes per graph
#define DEG   32
#define EPN   (NPG * DEG)         // 8192 regular edges per graph
#define ETOT  (EPN + NPG)         // + self loops
#define E_ALL (BGR * EPN)         // 2,097,152 edges per side
#define F0    7
#define F1    64
#define F2    32
#define NEG_SLOPE 0.2f
#define EPS_IN    1e-5f
#define S_SIZE ((size_t)BGR * NPG * NPG)   // 16,777,216
#define H_SIZE ((size_t)BGR * NPG * F2)    // 2,097,152
#define SPITCH 257                // LDS row pitch (floats) -> conflict-free cols

typedef __attribute__((ext_vector_type(16))) __bf16 v16bf;
typedef __attribute__((ext_vector_type(8)))  float  v8f;
typedef __attribute__((ext_vector_type(4)))  unsigned int u32x4;
typedef __attribute__((ext_vector_type(8)))  int i32x8;
typedef __attribute__((ext_vector_type(4)))  int i32x4;

#if defined(__HIP_DEVICE_COMPILE__) && __has_builtin(__builtin_amdgcn_tensor_load_to_lds)
#define HAVE_TDM 1
#else
#define HAVE_TDM 0
#endif

// ---------------------------------------------------------------------------
// TDM: 1-D DMA of `ndw` dwords from global to LDS (ISA 08_async_tensor §8).
// D# group0: count=1 | lds_addr | global_addr | type=2. group1: data_size=4B,
// 2D tile with tile_dim0=ndw, tile_dim1=1, tensor_dim0=ndw, stride=ndw.
// Issue from ONE wave; EXEC is ignored; tracked by TENSORcnt.
// ---------------------------------------------------------------------------
#if HAVE_TDM
__device__ __forceinline__ void tdm_load_1d_dwords(void* lds_ptr,
                                                   const void* gptr,
                                                   unsigned ndw) {
  // LDS aperture: flat addr[31:0] == wave-relative LDS byte address
  const unsigned lds_off = (unsigned)(unsigned long long)lds_ptr;
  const unsigned long long ga = (unsigned long long)gptr;
  u32x4 g0;
  g0[0] = 1u;                                    // count=1 (valid user D#)
  g0[1] = lds_off;                               // lds_addr[31:0]
  g0[2] = (unsigned)ga;                          // global_addr[31:0]
  g0[3] = (unsigned)((ga >> 32) & 0x01ffffffu)   // global_addr[56:32]
        | (2u << 30);                            // type=2 ("image")
  i32x8 g1;
  g1[0] = (int)(2u << 16);                       // data_size=2 (4 bytes), mask=0
  g1[1] = (int)((ndw & 0xffffu) << 16);          // tensor_dim0[15:0]
  g1[2] = (int)((ndw >> 16) & 0xffffu) | (1 << 16); // tensor_dim0[31:16] | tensor_dim1=1
  g1[3] = (int)((ndw & 0xffffu) << 16);          // tile_dim0 = ndw
  g1[4] = 1;                                     // tile_dim1 = 1 (tile_dim2=0)
  g1[5] = (int)ndw;                              // tensor_dim0_stride[31:0]
  g1[6] = 0;                                     // stride hi / dim1_stride lo
  g1[7] = 0;
  const i32x4 z4 = {0, 0, 0, 0};
#if defined(__clang_major__) && (__clang_major__ >= 23)
  const i32x8 z8 = {0, 0, 0, 0, 0, 0, 0, 0};
  __builtin_amdgcn_tensor_load_to_lds(g0, g1, z4, z4, z8, 0);
#else
  __builtin_amdgcn_tensor_load_to_lds(g0, g1, z4, z4, 0);
#endif
}
#endif

// ---------------------------------------------------------------------------
// helpers
// ---------------------------------------------------------------------------
__device__ __forceinline__ void atomicMaxF(float* addr, float v) {
  if (v >= 0.0f) atomicMax((int*)addr, __float_as_int(v));
  else           atomicMin((unsigned int*)addr, __float_as_uint(v));
}

template <int F>
__device__ __forceinline__ float edge_score(const float* __restrict__ xl,
                                            const float* __restrict__ xr,
                                            const float* __restrict__ att,
                                            int s, int d) {
  const float* a = xl + s * F;
  const float* r = xr + d * F;
  float acc = 0.0f;
#pragma unroll 8
  for (int f = 0; f < F; ++f) {
    float m = a[f] + r[f];
    m = (m > 0.0f) ? m : NEG_SLOPE * m;
    acc += m * att[f];
  }
  return acc;
}

// One GATv2 layer, fully LDS-resident for one graph.
template <int FIN, int FOUT>
__device__ void gat_layer(const float* __restrict__ sIn,   // [NPG][FIN]
                          const float* sWl, const float* sWr, const float* sAtt,
                          float* sXL, float* sXR, float* sOut,
                          float* sEmx, float* sDen,
                          const long long* __restrict__ srcp,
                          const long long* __restrict__ dstp,
                          int tid) {
  // node transforms: xl = x@Wl, xr = x@Wr
  for (int idx = tid; idx < NPG * FOUT; idx += 256) {
    const int n = idx / FOUT, f = idx % FOUT;
    const float* xn = sIn + n * FIN;
    float al = 0.0f, ar = 0.0f;
#pragma unroll 8
    for (int k = 0; k < FIN; ++k) {
      const float xv = xn[k];
      al += xv * sWl[k * FOUT + f];
      ar += xv * sWr[k * FOUT + f];
    }
    sXL[idx] = al;
    sXR[idx] = ar;
  }
  for (int n = tid; n < NPG; n += 256) {
    sEmx[n] = __int_as_float(0xff800000);  // -inf
    sDen[n] = 0.0f;
  }
  for (int i = tid; i < NPG * FOUT; i += 256) sOut[i] = 0.0f;
  __syncthreads();

  // pass A: segment max of attention logits over dst (LDS atomic)
  for (int e = tid; e < ETOT; e += 256) {
    int s, d;
    if (e < EPN) { s = (int)srcp[e] & (NPG - 1); d = (int)dstp[e] & (NPG - 1); }
    else         { s = d = e - EPN; }
    atomicMaxF(&sEmx[d], edge_score<FOUT>(sXL, sXR, sAtt, s, d));
  }
  __syncthreads();

  // pass B: softmax denominator
  for (int e = tid; e < ETOT; e += 256) {
    int s, d;
    if (e < EPN) { s = (int)srcp[e] & (NPG - 1); d = (int)dstp[e] & (NPG - 1); }
    else         { s = d = e - EPN; }
    const float ev = edge_score<FOUT>(sXL, sXR, sAtt, s, d);
    atomicAdd(&sDen[d], expf(ev - sEmx[d]));
  }
  __syncthreads();

  // pass C: weighted aggregation out[dst] += alpha * xl[src]
  for (int e = tid; e < ETOT; e += 256) {
    int s, d;
    if (e < EPN) { s = (int)srcp[e] & (NPG - 1); d = (int)dstp[e] & (NPG - 1); }
    else         { s = d = e - EPN; }
    const float ev = edge_score<FOUT>(sXL, sXR, sAtt, s, d);
    const float alpha = expf(ev - sEmx[d]) / (sDen[d] + 1e-16f);
    const float* xs = sXL + s * FOUT;
    float* od = sOut + d * FOUT;
#pragma unroll 8
    for (int f = 0; f < FOUT; ++f) atomicAdd(&od[f], alpha * xs[f]);
  }
  __syncthreads();
}

// ---------------------------------------------------------------------------
// Kernel 1: fused two-layer GATv2 encoder. One workgroup per (graph, side).
// ---------------------------------------------------------------------------
// dynamic LDS layout (floats):
//  Wl1 448 | Wr1 448 | att1 64 | b1 64 | Wl2 2048 | Wr2 2048 | att2 32 | b2 32
//  X 1792 | XL1 16384 | XR1 16384 | H 16384 | XL2 8192 | XR2 8192 | O2 8192
//  Emx 256 | Den 256           => 81216 floats = 324,864 B (<= 320 KB WGP LDS)
#define ENC_SMEM_BYTES (81216u * 4u)

__global__ __launch_bounds__(256) void encoder_kernel(
    const float* __restrict__ x1, const float* __restrict__ x2,
    const float* __restrict__ Wl1, const float* __restrict__ Wr1,
    const float* __restrict__ att1, const float* __restrict__ b1,
    const float* __restrict__ Wl2, const float* __restrict__ Wr2,
    const float* __restrict__ att2, const float* __restrict__ b2,
    const long long* __restrict__ e1, const long long* __restrict__ e2,
    float* __restrict__ dout, __bf16* __restrict__ hbf_ws) {
  extern __shared__ char smem[];
  float* sWl1 = (float*)smem;
  float* sWr1 = sWl1 + 448;
  float* sAtt1 = sWr1 + 448;
  float* sB1 = sAtt1 + 64;
  float* sWl2 = sB1 + 64;
  float* sWr2 = sWl2 + 2048;
  float* sAtt2 = sWr2 + 2048;
  float* sB2 = sAtt2 + 32;
  float* sX = sB2 + 32;
  float* sXL1 = sX + 1792;
  float* sXR1 = sXL1 + 16384;
  float* sH = sXR1 + 16384;
  float* sXL2 = sH + 16384;
  float* sXR2 = sXL2 + 8192;
  float* sO2 = sXR2 + 8192;
  float* sEmx = sO2 + 8192;
  float* sDen = sEmx + 256;

  const int tid = threadIdx.x;
  const int wave = tid >> 5;
  const int side = blockIdx.x >> 8;   // 0 -> graph set 1, 1 -> graph set 2
  const int b = blockIdx.x & 255;

  const float* x = side ? x2 : x1;
  const long long* ee = side ? e2 : e1;
  const long long* srcp = ee + (size_t)b * EPN;
  const long long* dstp = ee + (size_t)E_ALL + (size_t)b * EPN;

#if HAVE_TDM
  if (wave == 0)  // async DMA of this graph's node features into LDS
    tdm_load_1d_dwords(sX, x + (size_t)b * NPG * F0, NPG * F0);
#endif
  for (int i = tid; i < F0 * F1; i += 256) { sWl1[i] = Wl1[i]; sWr1[i] = Wr1[i]; }
  for (int i = tid; i < F1; i += 256)      { sAtt1[i] = att1[i]; sB1[i] = b1[i]; }
  for (int i = tid; i < F1 * F2; i += 256) { sWl2[i] = Wl2[i]; sWr2[i] = Wr2[i]; }
  for (int i = tid; i < F2; i += 256)      { sAtt2[i] = att2[i]; sB2[i] = b2[i]; }
#if HAVE_TDM
  if (wave == 0) __builtin_amdgcn_s_wait_tensorcnt(0);
#else
  for (int i = tid; i < NPG * F0; i += 256) sX[i] = x[(size_t)b * NPG * F0 + i];
#endif
  __syncthreads();

  gat_layer<F0, F1>(sX, sWl1, sWr1, sAtt1, sXL1, sXR1, sH, sEmx, sDen, srcp, dstp, tid);

  // h = relu(out1 + b1)
  for (int i = tid; i < NPG * F1; i += 256) {
    const float v = sH[i] + sB1[i & (F1 - 1)];
    sH[i] = (v > 0.0f) ? v : 0.0f;
  }
  __syncthreads();

  gat_layer<F1, F2>(sH, sWl2, sWr2, sAtt2, sXL2, sXR2, sO2, sEmx, sDen, srcp, dstp, tid);

  // final node embedding h = out2 + b2: f32 -> d_out (result), bf16 -> ws
  float* hout = dout + S_SIZE + (size_t)side * H_SIZE + (size_t)b * NPG * F2;
  __bf16* hbf = hbf_ws + (size_t)side * H_SIZE + (size_t)b * NPG * F2;
  for (int i = tid; i < NPG * F2; i += 256) {
    const float v = sO2[i] + sB2[i & (F2 - 1)];
    hout[i] = v;
    hbf[i] = (__bf16)v;
  }
}

// ---------------------------------------------------------------------------
// Kernel 2: per-graph similarity (bf16 WMMA) + InstanceNorm + 10 Sinkhorn
// iterations + exp, fully LDS-resident. One workgroup (32 waves) per graph.
// bf16 embeddings arrive via TENSOR_LOAD_TO_LDS DMA (TDM) from workspace.
// ---------------------------------------------------------------------------
// LDS: sim 256x257 f32 (263,168 B) + h1/h2 bf16 (2x16,384 B) + scratch
#define SIM_SMEM_BYTES (263168u + 16384u + 16384u + 512u)

__global__ __launch_bounds__(1024) void sim_sinkhorn_kernel(
    float* __restrict__ dout, const __bf16* __restrict__ hbf_ws,
    const float* __restrict__ gammaP, const float* __restrict__ betaP) {
  extern __shared__ char smem[];
  float* sS = (float*)smem;                       // [256][SPITCH]
  __bf16* sH1 = (__bf16*)(sS + NPG * SPITCH);     // [256][32]
  __bf16* sH2 = sH1 + NPG * F2;                   // [256][32]
  float* sRed = (float*)(sH2 + NPG * F2);         // 66 floats

  const int tid = threadIdx.x;
  const int lane = tid & 31;
  const int wave = tid >> 5;
  const int b = blockIdx.x;

  const __bf16* h1 = hbf_ws + (size_t)b * NPG * F2;
  const __bf16* h2 = hbf_ws + H_SIZE + (size_t)b * NPG * F2;
#if HAVE_TDM
  if (wave == 0) {  // two 16 KB DMA tiles straight into their LDS slots
    tdm_load_1d_dwords(sH1, h1, NPG * F2 / 2);
    tdm_load_1d_dwords(sH2, h2, NPG * F2 / 2);
    __builtin_amdgcn_s_wait_tensorcnt(0);
  }
#else
  {
    const uint4* g1 = (const uint4*)h1;
    const uint4* g2 = (const uint4*)h2;
    uint4* d1 = (uint4*)sH1;
    uint4* d2 = (uint4*)sH2;
    for (int i = tid; i < NPG * F2 * 2 / 16; i += 1024) { d1[i] = g1[i]; d2[i] = g2[i]; }
  }
#endif
  __syncthreads();

  // --- sim = h1 @ h2^T via v_wmma_f32_16x16x32_bf16 (K=32 == d, one WMMA/tile)
  // A frag (16x32, MxK): lanes 0-15: M=lane, K in {0..7,16..23}; lanes 16-31: K in {8..15,24..31}
  // B frag (32x16, KxN): lanes 0-15: N=lane, K in {0..7,16..23}; lanes 16-31: N=lane-16, K in {8..15,24..31}
  // D frag: element v at (M = 16*tm + 8*half + v, N = 16*tn + lane%16)
  {
    const int lrow = lane & 15;
    const int half = lane >> 4;
    const int tm = wave >> 1;            // each wave's 8 tiles share one tm
    const int tn0 = (wave & 1) * 8;
    v16bf A;                             // loop-invariant A fragment
    const __bf16* ra = sH1 + (tm * 16 + lrow) * F2;
    const int kA = half * 8;
#pragma unroll
    for (int i = 0; i < 8; ++i) { A[i] = ra[kA + i]; A[8 + i] = ra[kA + 16 + i]; }
    const int kB = half * 8;
    for (int tn = tn0; tn < tn0 + 8; ++tn) {
      v16bf Bf;
      const __bf16* rb = sH2 + (tn * 16 + lrow) * F2;
#pragma unroll
      for (int i = 0; i < 16; ++i) Bf[i] = rb[kB + (i & 7) + ((i >> 3) << 4)];
      v8f C = {};
      C = __builtin_amdgcn_wmma_f32_16x16x32_bf16(false, A, false, Bf,
                                                  (short)0, C, false, false);
      const int n = tn * 16 + lrow;
      const int mb = tm * 16 + half * 8;
#pragma unroll
      for (int v = 0; v < 8; ++v) sS[(mb + v) * SPITCH + n] = C[v];
    }
  }
  __syncthreads();

  // --- InstanceNorm2d statistics (biased var) over 256x256
  float s = 0.0f, ss = 0.0f;
  for (int i = tid; i < NPG * NPG; i += 1024) {
    const float v = sS[(i >> 8) * SPITCH + (i & 255)];
    s += v; ss += v * v;
  }
#pragma unroll
  for (int off = 16; off; off >>= 1) { s += __shfl_xor(s, off); ss += __shfl_xor(ss, off); }
  if (lane == 0) { sRed[wave] = s; sRed[32 + wave] = ss; }
  __syncthreads();
  if (wave == 0) {
    float a = sRed[lane], c = sRed[32 + lane];
#pragma unroll
    for (int off = 16; off; off >>= 1) { a += __shfl_xor(a, off); c += __shfl_xor(c, off); }
    if (lane == 0) {
      const float mean = a * (1.0f / 65536.0f);
      const float var = c * (1.0f / 65536.0f) - mean * mean;
      sRed[64] = mean;
      sRed[65] = rsqrtf(var + EPS_IN) * gammaP[0];
    }
  }
  __syncthreads();
  const float mean = sRed[64], scale = sRed[65], beta = betaP[0];
  // log_s = sim_n / TAU  (TAU == 1)
  for (int i = tid; i < NPG * NPG; i += 1024) {
    float* p = &sS[(i >> 8) * SPITCH + (i & 255)];
    *p = (*p - mean) * scale + beta;
  }
  __syncthreads();

  // --- 10 log-space Sinkhorn iterations (even: rows/axis2, odd: cols/axis1)
  for (int it = 0; it < 10; ++it) {
    if ((it & 1) == 0) {
      for (int r = wave; r < NPG; r += 32) {        // one wave owns a row
        float* row = sS + r * SPITCH;
        float m = -3.4e38f;
#pragma unroll
        for (int k = 0; k < 8; ++k) m = fmaxf(m, row[lane + 32 * k]);
#pragma unroll
        for (int off = 16; off; off >>= 1) m = fmaxf(m, __shfl_xor(m, off));
        float sm = 0.0f;
#pragma unroll
        for (int k = 0; k < 8; ++k) sm += expf(row[lane + 32 * k] - m);
#pragma unroll
        for (int off = 16; off; off >>= 1) sm += __shfl_xor(sm, off);
        const float lse = m + logf(sm);
#pragma unroll
        for (int k = 0; k < 8; ++k) row[lane + 32 * k] -= lse;
      }
    } else {
      for (int c = wave; c < NPG; c += 32) {        // one wave owns a column
        float* col = sS + c;                        // pitch 257 -> conflict free
        float m = -3.4e38f;
#pragma unroll
        for (int k = 0; k < 8; ++k) m = fmaxf(m, col[(lane + 32 * k) * SPITCH]);
#pragma unroll
        for (int off = 16; off; off >>= 1) m = fmaxf(m, __shfl_xor(m, off));
        float sm = 0.0f;
#pragma unroll
        for (int k = 0; k < 8; ++k) sm += expf(col[(lane + 32 * k) * SPITCH] - m);
#pragma unroll
        for (int off = 16; off; off >>= 1) sm += __shfl_xor(sm, off);
        const float lse = m + logf(sm);
#pragma unroll
        for (int k = 0; k < 8; ++k) col[(lane + 32 * k) * SPITCH] -= lse;
      }
    }
    __syncthreads();
  }

  // --- S = exp(log_s) -> d_out
  float* Sout = dout + (size_t)b * NPG * NPG;
  for (int i = tid; i < NPG * NPG; i += 1024)
    Sout[i] = expf(sS[(i >> 8) * SPITCH + (i & 255)]);
}

// ---------------------------------------------------------------------------
// launcher
// ---------------------------------------------------------------------------
extern "C" void kernel_launch(void* const* d_in, const int* in_sizes, int n_in,
                              void* d_out, int out_size, void* d_ws, size_t ws_size,
                              hipStream_t stream) {
  (void)in_sizes; (void)n_in; (void)out_size; (void)ws_size;

  const float* x1   = (const float*)d_in[0];
  const float* x2   = (const float*)d_in[1];
  const float* Wl1  = (const float*)d_in[2];
  const float* Wr1  = (const float*)d_in[3];
  const float* att1 = (const float*)d_in[4];
  const float* b1   = (const float*)d_in[5];
  const float* Wl2  = (const float*)d_in[6];
  const float* Wr2  = (const float*)d_in[7];
  const float* att2 = (const float*)d_in[8];
  const float* b2   = (const float*)d_in[9];
  const float* gamma = (const float*)d_in[10];
  const float* beta  = (const float*)d_in[11];
  const long long* e1 = (const long long*)d_in[12];  // int64 per reference
  const long long* e2 = (const long long*)d_in[13];
  float* out = (float*)d_out;
  __bf16* hbf_ws = (__bf16*)d_ws;   // 2 * H_SIZE bf16 = 8.4 MB staging

  // allow >64KB dynamic LDS (CDNA5 WGP: 320 KB); idempotent, capture-safe
  hipFuncSetAttribute((const void*)encoder_kernel,
                      hipFuncAttributeMaxDynamicSharedMemorySize, ENC_SMEM_BYTES);
  hipFuncSetAttribute((const void*)sim_sinkhorn_kernel,
                      hipFuncAttributeMaxDynamicSharedMemorySize, SIM_SMEM_BYTES);

  encoder_kernel<<<dim3(2 * BGR), dim3(256), ENC_SMEM_BYTES, stream>>>(
      x1, x2, Wl1, Wr1, att1, b1, Wl2, Wr2, att2, b2, e1, e2, out, hbf_ws);
  sim_sinkhorn_kernel<<<dim3(BGR), dim3(1024), SIM_SMEM_BYTES, stream>>>(
      out, hbf_ws, gamma, beta);
}
